// GVPDualBranchConvLayer_19567871001009
// MI455X (gfx1250) — compile-verified, matrix operand order checked by reading, not run on hardware
//
#include <hip/hip_runtime.h>
#include <hip/hip_bf16.h>
#include <math.h>

// ---------------------------------------------------------------------------
// GVP attention conv layer for MI455X (gfx1250, wave32, WMMA bf16->f32).
// N=10000 nodes, E=320000 edges, NC=1, SI=SO=128, VI=VO=16, SE=32, VE=1,
// HEADS=4 (HS=32, HV=4).
// ---------------------------------------------------------------------------

#define N_NODES 10000
#define N_EDGES 320000
#define EPS_GVP 1e-8f

typedef __attribute__((ext_vector_type(16))) __bf16 v16bf;
typedef __attribute__((ext_vector_type(8)))  float  v8f;

union FragU { v16bf v; unsigned u[8]; };
union BF2  { __bf16 b[2]; unsigned u; };

__device__ __forceinline__ float sigmoidf(float x) { return 1.0f / (1.0f + __expf(-x)); }
__device__ __forceinline__ float siluf(float x)    { return x * sigmoidf(x); }

__device__ __forceinline__ v8f wmma_bf16(v16bf a, v16bf b, v8f c) {
  return __builtin_amdgcn_wmma_f32_16x16x32_bf16(false, a, false, b, (short)0, c,
                                                 false, false);
}

// 16x32 bf16 fragment loader (serves A row-major and B staged N-major).
// CDNA5 layout: lanes 0-15 -> row=lane, element pairs cover K {0..7,16..23};
// lanes 16-31 -> row=lane-16, K {8..15,24..31}. base points at (row0, k-window
// start); ldk = row stride in elements (even); kmax = valid K in this 32-wide
// window (even) -> zero-fill beyond it (K padding without storing pad columns).
__device__ __forceinline__ v16bf load_frag(const __bf16* base, int ldk, int kmax) {
  const int lane = threadIdx.x & 31;
  const int row  = lane & 15;
  const int hi   = lane >> 4;
  const __bf16* r = base + row * ldk;
  FragU f;
#pragma unroll
  for (int j = 0; j < 8; ++j) {
    const int k0 = ((j & 4) << 2) + (hi << 3) + ((j & 3) << 1);
    f.u[j] = (k0 < kmax) ? *(const unsigned*)(r + k0) : 0u;
  }
  return f.v;
}

// Fragment loader that reads f32 from LDS, applies sigmoid, packs bf16 pairs.
// Used for the vector-gate GEMM A operand (A = sigmoid(s_pre)).
__device__ __forceinline__ v16bf load_frag_sig_f32(const float* base, int ldk) {
  const int lane = threadIdx.x & 31;
  const int row  = lane & 15;
  const int hi   = lane >> 4;
  const float* r = base + row * ldk;
  FragU f;
#pragma unroll
  for (int j = 0; j < 8; ++j) {
    const int k0 = ((j & 4) << 2) + (hi << 3) + ((j & 3) << 1);
    BF2 p;
    p.b[0] = (__bf16)sigmoidf(r[k0]);
    p.b[1] = (__bf16)sigmoidf(r[k0 + 1]);
    f.u[j] = p.u;
  }
  return f.v;
}

__device__ __forceinline__ void atomicMaxF(float* addr, float val) {
  int* ia = (int*)addr;
  int cur = __float_as_int(*addr);
  while (__int_as_float(cur) < val) {
    int old = atomicCAS(ia, cur, __float_as_int(val));
    if (old == cur) break;
    cur = old;
  }
}

// ---------------------------------------------------------------------------
// Node GVP (si=128, vi=16 -> so=128, vo=16), vector_gate, silu/sigmoid acts.
// One block = 16 rows, 4 waves. s_pre GEMM: K=144 (pad->5 ksteps), N=128
// (2 tiles/wave). Gate GEMM: K=128, N=16 (wave 0).
// ---------------------------------------------------------------------------
__global__ __launch_bounds__(128)
void node_gvp_kernel(const float* __restrict__ in_s,  // [R,128]
                     const float* __restrict__ in_v,  // [R,16,3]
                     const float* __restrict__ wh,    // [16,16]
                     const float* __restrict__ ws,    // [128,144]
                     const float* __restrict__ bs,    // [128]
                     const float* __restrict__ wv,    // [16,16]
                     const float* __restrict__ wsv,   // [16,128]
                     const float* __restrict__ bsv,   // [16]
                     float* __restrict__ out_s,       // [R,128]
                     float* __restrict__ out_v,       // [R,16,3]
                     int rows)
{
  __shared__ __bf16 sA[16][144];     // [s | vn], K=144
  __shared__ __bf16 sB[128][144];    // ws, N-major
  __shared__ __bf16 sBg[16][128];    // wsv, N-major
  __shared__ float  sVh[16][3][16];
  __shared__ float  sSP[16][128];
  __shared__ float  sGate[16][16];
  __shared__ float  sWh[16][16], sWv[16][16], sBs[128], sBsv[16];

  const int tid  = threadIdx.x;
  const int lane = tid & 31, wave = tid >> 5;
  const int r0   = blockIdx.x * 16;

  for (int i = tid; i < 256; i += 128) { sWh[i >> 4][i & 15] = wh[i]; sWv[i >> 4][i & 15] = wv[i]; }
  if (tid < 128) sBs[tid] = bs[tid];
  if (tid < 16)  sBsv[tid] = bsv[tid];
  for (int i = tid; i < 128 * 144; i += 128) sB[i / 144][i % 144] = (__bf16)ws[i];
  for (int i = tid; i < 16 * 128; i += 128)  sBg[i >> 7][i & 127] = (__bf16)wsv[i];
  __syncthreads();

  // vh[m][d][h] = sum_c in_v[m][c][d] * wh[h][c]
  for (int i = tid; i < 16 * 48; i += 128) {
    int m = i / 48, rem = i % 48, d = rem / 16, h = rem % 16;
    float acc = 0.f;
    if (r0 + m < rows) {
      const float* vp = in_v + (size_t)(r0 + m) * 48;
#pragma unroll
      for (int c = 0; c < 16; ++c) acc += vp[c * 3 + d] * sWh[h][c];
    }
    sVh[m][d][h] = acc;
  }
  __syncthreads();
  for (int i = tid; i < 256; i += 128) {
    int m = i >> 4, h = i & 15;
    float s2 = sVh[m][0][h] * sVh[m][0][h] + sVh[m][1][h] * sVh[m][1][h] +
               sVh[m][2][h] * sVh[m][2][h];
    sA[m][128 + h] = (__bf16)sqrtf(fmaxf(s2, EPS_GVP));
  }
  for (int i = tid; i < 16 * 128; i += 128) {
    int m = i >> 7, c = i & 127;
    sA[m][c] = (r0 + m < rows) ? (__bf16)in_s[(size_t)(r0 + m) * 128 + c] : (__bf16)0.f;
  }
  __syncthreads();

  // s_pre: each wave -> 2 output tiles of 16
  {
    const int t0 = wave * 2, t1 = wave * 2 + 1;
    v8f acc0 = {}, acc1 = {};
#pragma unroll
    for (int kk = 0; kk < 144; kk += 32) {
      const int kmax = (144 - kk > 32) ? 32 : (144 - kk);
      v16bf a  = load_frag(&sA[0][0] + kk, 144, kmax);
      v16bf b0 = load_frag(&sB[t0 * 16][0] + kk, 144, kmax);
      v16bf b1 = load_frag(&sB[t1 * 16][0] + kk, 144, kmax);
      acc0 = wmma_bf16(a, b0, acc0);
      acc1 = wmma_bf16(a, b1, acc1);
    }
    const int n = lane & 15, hi = lane >> 4;
#pragma unroll
    for (int j = 0; j < 8; ++j) {
      const int m = j + hi * 8;
      sSP[m][t0 * 16 + n] = acc0[j] + sBs[t0 * 16 + n];
      sSP[m][t1 * 16 + n] = acc1[j] + sBs[t1 * 16 + n];
    }
  }
  __syncthreads();

  // gate = sigmoid(s_pre) @ wsv^T + bsv  (wave 0, K=128)
  if (wave == 0) {
    v8f g = {};
#pragma unroll
    for (int kk = 0; kk < 128; kk += 32) {
      v16bf a = load_frag_sig_f32(&sSP[0][0] + kk, 128);
      v16bf b = load_frag(&sBg[0][0] + kk, 128, 32);
      g = wmma_bf16(a, b, g);
    }
    const int n = lane & 15, hi = lane >> 4;
#pragma unroll
    for (int j = 0; j < 8; ++j) sGate[j + hi * 8][n] = g[j] + sBsv[n];
  }
  __syncthreads();

  for (int i = tid; i < 16 * 128; i += 128) {
    int m = i >> 7, c = i & 127;
    if (r0 + m < rows) out_s[(size_t)(r0 + m) * 128 + c] = siluf(sSP[m][c]);
  }
  for (int i = tid; i < 16 * 48; i += 128) {
    int m = i / 48, rem = i % 48, o = rem / 3, d = rem % 3;
    float acc = 0.f;
#pragma unroll
    for (int h = 0; h < 16; ++h) acc += sVh[m][d][h] * sWv[o][h];
    if (r0 + m < rows)
      out_v[(size_t)(r0 + m) * 48 + o * 3 + d] = acc * sigmoidf(sGate[m][o]);
  }
}

// ---------------------------------------------------------------------------
// Fused edge pass. MODE 0: ek GVP + k_b + q gather + attention energy +
// segment max. MODE 1: ev GVP + v_b + alpha-weighted atomic aggregation.
// One block = 64 edges, 4 waves (16 edges/wave). s_pre: K=48 (2 ksteps),
// N=128 (8 tiles). Gate: K=128 (4 ksteps), N=16. Since VE=1,
// vout_e[o][d] = edge_v[d] * dot(wv[o,:], wh[:,0]) (precomputed sWvWh).
// ---------------------------------------------------------------------------
template <int MODE>
__global__ __launch_bounds__(128)
void edge_kernel(const float* __restrict__ edge_s,  // [E,32]
                 const float* __restrict__ edge_v,  // [E,1,3]
                 const int* __restrict__ src, const int* __restrict__ dst,
                 const float* __restrict__ wh,   // [16,1]
                 const float* __restrict__ ws,   // [128,48]
                 const float* __restrict__ bs,   // [128]
                 const float* __restrict__ wv,   // [16,16]
                 const float* __restrict__ wsv,  // [16,128]
                 const float* __restrict__ bsv,  // [16]
                 const float* __restrict__ node_s,  // ks (M0) / vs (M1)
                 const float* __restrict__ node_v,  // kv (M0) / vv (M1)
                 const float* __restrict__ qs, const float* __restrict__ qv,
                 const float* __restrict__ whA,  // [8,8]
                 const float* __restrict__ wsA,  // [72]
                 const float* __restrict__ bsA,  // [1]
                 float* __restrict__ energy,     // M0: out e; M1: in exp
                 float* __restrict__ nmax,       // M0
                 const float* __restrict__ nden, // M1
                 float* __restrict__ aggs, float* __restrict__ aggv)  // M1
{
  __shared__ __bf16 sBk[128][48];
  __shared__ __bf16 sBg[16][128];
  __shared__ __bf16 sA[4][16][48];
  __shared__ float  sSP[4][16][128];
  __shared__ float  sGate[4][16][16];
  __shared__ int    sSrc[4][16], sDst[4][16];
  __shared__ float  sEV[4][16][3];
  __shared__ float  sAlpha[4][16][4];
  __shared__ float  sBsE[128], sBsvE[16], sWhE[16], sWvWh[16];
  __shared__ float  sWsA[72], sWhA[8][8], sBsA[1];

  const int tid  = threadIdx.x;
  const int lane = tid & 31, wave = tid >> 5;
  const int e0   = blockIdx.x * 64;

  // ---- Phase A: stage weights + per-edge meta + edge scalar features ----
  for (int i = tid; i < 128 * 48; i += 128) sBk[i / 48][i % 48] = (__bf16)ws[i];
  for (int i = tid; i < 16 * 128; i += 128) sBg[i >> 7][i & 127] = (__bf16)wsv[i];
  if (tid < 128) sBsE[tid] = bs[tid];
  if (tid < 16) {
    sBsvE[tid] = bsv[tid];
    sWhE[tid]  = wh[tid];
    float a = 0.f;
#pragma unroll
    for (int h = 0; h < 16; ++h) a += wv[tid * 16 + h] * wh[h];
    sWvWh[tid] = a;
  }
  if (MODE == 0) {
    if (tid < 72) sWsA[tid] = wsA[tid];
    if (tid < 64) sWhA[tid >> 3][tid & 7] = whA[tid];
    if (tid == 0) sBsA[0] = bsA[0];
  }
  if (lane < 16) {
    const int e = e0 + wave * 16 + lane;
    sSrc[wave][lane] = src[e];
    sDst[wave][lane] = dst[e];
    sEV[wave][lane][0] = edge_v[(size_t)e * 3 + 0];
    sEV[wave][lane][1] = edge_v[(size_t)e * 3 + 1];
    sEV[wave][lane][2] = edge_v[(size_t)e * 3 + 2];
  }
  for (int i = lane; i < 512; i += 32) {
    int m = i >> 5, c = i & 31;
    sA[wave][m][c] = (__bf16)edge_s[(size_t)(e0 + wave * 16 + m) * 32 + c];
  }
  __syncthreads();

  // ---- Phase B: vn (|wh[h]| * |v|), plus alpha staging in MODE 1 ----
  for (int i = lane; i < 256; i += 32) {
    int m = i >> 4, h = i & 15;
    float n2 = sEV[wave][m][0] * sEV[wave][m][0] + sEV[wave][m][1] * sEV[wave][m][1] +
               sEV[wave][m][2] * sEV[wave][m][2];
    float t = sWhE[h] * sWhE[h] * n2;
    sA[wave][m][32 + h] = (__bf16)sqrtf(fmaxf(t, EPS_GVP));
  }
  if (MODE == 1) {
    for (int t = tid; t < 256; t += 128) {
      int le = t >> 2, h = t & 3, w2 = le >> 4, m = le & 15;
      int d = sDst[w2][m];
      sAlpha[w2][m][h] =
          energy[(size_t)(e0 + le) * 4 + h] / (nden[(size_t)d * 4 + h] + 1e-16f);
    }
  }
  __syncthreads();

  // ---- Phase C: s_pre GEMM (per-wave 16 edges, 8 N-tiles, 2 ksteps) ----
  {
    v8f acc[8] = {};
    const __bf16* Aw = &sA[wave][0][0];
#pragma unroll
    for (int kk = 0; kk < 48; kk += 32) {
      const int kmax = (48 - kk > 32) ? 32 : (48 - kk);
      v16bf a = load_frag(Aw + kk, 48, kmax);
#pragma unroll
      for (int t = 0; t < 8; ++t) {
        v16bf b = load_frag(&sBk[t * 16][0] + kk, 48, kmax);
        acc[t] = wmma_bf16(a, b, acc[t]);
      }
    }
    const int n = lane & 15, hi = lane >> 4;
#pragma unroll
    for (int t = 0; t < 8; ++t)
#pragma unroll
      for (int j = 0; j < 8; ++j)
        sSP[wave][j + hi * 8][t * 16 + n] = acc[t][j] + sBsE[t * 16 + n];
  }
  __syncthreads();

  // ---- Phase D: gate GEMM, then overwrite sSP with node_s[src] + silu(sp) ----
  {
    v8f g = {};
#pragma unroll
    for (int kk = 0; kk < 128; kk += 32) {
      v16bf a = load_frag_sig_f32(&sSP[wave][0][0] + kk, 128);
      v16bf b = load_frag(&sBg[0][0] + kk, 128, 32);
      g = wmma_bf16(a, b, g);
    }
    const int n = lane & 15, hi = lane >> 4;
#pragma unroll
    for (int j = 0; j < 8; ++j) sGate[wave][j + hi * 8][n] = g[j] + sBsvE[n];
    for (int i = lane; i < 2048; i += 32) {
      int m = i >> 7, c = i & 127;
      float sp = sSP[wave][m][c];
      sSP[wave][m][c] = node_s[(size_t)sSrc[wave][m] * 128 + c] + siluf(sp);
    }
  }
  __syncthreads();

  // ---- Phase E ----
  if (MODE == 0) {
    // attention energy per (edge, head): 64 edges * 4 heads = 256 tasks
    for (int t = tid; t < 256; t += 128) {
      const int le = t >> 2, h = t & 3;
      const int w2 = le >> 4, m = le & 15;
      const int d = sDst[w2][m], s = sSrc[w2][m];
      float kbv[4][3], qiv[4][3];
#pragma unroll
      for (int oo = 0; oo < 4; ++oo) {
        const int o = h * 4 + oo;
        const float gv = sigmoidf(sGate[w2][m][o]);
#pragma unroll
        for (int dd = 0; dd < 3; ++dd) {
          kbv[oo][dd] = node_v[(size_t)s * 48 + o * 3 + dd] +
                        sEV[w2][m][dd] * sWvWh[o] * gv;
          qiv[oo][dd] = qv[(size_t)d * 48 + o * 3 + dd];
        }
      }
      float e = sBsA[0];
      const float* qsd = qs + (size_t)d * 128 + h * 32;
#pragma unroll
      for (int i2 = 0; i2 < 32; ++i2) e += sWsA[i2] * qsd[i2];
#pragma unroll
      for (int i2 = 0; i2 < 32; ++i2) e += sWsA[32 + i2] * sSP[w2][m][h * 32 + i2];
#pragma unroll
      for (int j2 = 0; j2 < 8; ++j2) {
        float n2 = 0.f;
#pragma unroll
        for (int dd = 0; dd < 3; ++dd) {
          float vh = 0.f;
#pragma unroll
          for (int c = 0; c < 8; ++c) {
            float av = (c < 4) ? qiv[c][dd] : kbv[c - 4][dd];
            vh += av * sWhA[j2][c];
          }
          n2 += vh * vh;
        }
        e += sWsA[64 + j2] * sqrtf(fmaxf(n2, EPS_GVP));
      }
      energy[(size_t)(e0 + le) * 4 + h] = e;
      atomicMaxF(&nmax[(size_t)d * 4 + h], e);
    }
  } else {
    // scalar messages: 64 edges * 128 channels
    for (int i = tid; i < 64 * 128; i += 128) {
      int le = i >> 7, c = i & 127, w2 = le >> 4, m = le & 15;
      atomicAdd(&aggs[(size_t)sDst[w2][m] * 128 + c],
                sAlpha[w2][m][c >> 5] * sSP[w2][m][c]);
    }
    // vector messages: 64 edges * 16 channels * 3 dims
    for (int i = tid; i < 64 * 48; i += 128) {
      int le = i / 48, rem = i % 48, o = rem / 3, dd = rem % 3;
      int w2 = le >> 4, m = le & 15;
      int s = sSrc[w2][m];
      float vb = node_v[(size_t)s * 48 + o * 3 + dd] +
                 sEV[w2][m][dd] * sWvWh[o] * sigmoidf(sGate[w2][m][o]);
      atomicAdd(&aggv[(size_t)sDst[w2][m] * 48 + o * 3 + dd],
                sAlpha[w2][m][o >> 2] * vb);
    }
  }
}

// ---------------------------------------------------------------------------
__global__ void init_kernel(float* nmax, float* nden, float* aggs, float* aggv) {
  const int i = blockIdx.x * blockDim.x + threadIdx.x;
  if (i < N_NODES * 4) { nmax[i] = -3.0e38f; nden[i] = 0.f; }
  if (i < N_NODES * 128) aggs[i] = 0.f;
  if (i < N_NODES * 48)  aggv[i] = 0.f;
}

__global__ void softmax_kernel(const int* __restrict__ dst, float* __restrict__ energy,
                               const float* __restrict__ nmax, float* __restrict__ nden) {
  const int i = blockIdx.x * blockDim.x + threadIdx.x;
  if (i >= N_EDGES * 4) return;
  const int e = i >> 2, h = i & 3;
  const int d = dst[e];
  const float ex = __expf(energy[i] - nmax[(size_t)d * 4 + h]);
  energy[i] = ex;
  atomicAdd(&nden[(size_t)d * 4 + h], ex);
}

// ---------------------------------------------------------------------------
// Workspace layout (floats)
// ---------------------------------------------------------------------------
constexpr size_t OFF_QS   = 0;
constexpr size_t OFF_QV   = OFF_QS + (size_t)N_NODES * 128;
constexpr size_t OFF_KS   = OFF_QV + (size_t)N_NODES * 48;
constexpr size_t OFF_KV   = OFF_KS + (size_t)N_NODES * 128;
constexpr size_t OFF_VS   = OFF_KV + (size_t)N_NODES * 48;
constexpr size_t OFF_VV   = OFF_VS + (size_t)N_NODES * 128;
constexpr size_t OFF_EN   = OFF_VV + (size_t)N_NODES * 48;
constexpr size_t OFF_NMAX = OFF_EN + (size_t)N_EDGES * 4;
constexpr size_t OFF_NDEN = OFF_NMAX + (size_t)N_NODES * 4;
constexpr size_t OFF_AGGS = OFF_NDEN + (size_t)N_NODES * 4;
constexpr size_t OFF_AGGV = OFF_AGGS + (size_t)N_NODES * 128;

struct P6 { const float *wh, *ws, *bs, *wv, *wsv, *bsv; };

extern "C" void kernel_launch(void* const* d_in, const int* in_sizes, int n_in,
                              void* d_out, int out_size, void* d_ws, size_t ws_size,
                              hipStream_t stream) {
  (void)n_in; (void)out_size; (void)ws_size;
  auto F = [&](int i) { return (const float*)d_in[i]; };

  const float *x_s, *x_v, *edge_s, *edge_v;
  const int* eidx;
  P6 q, k, v, ek, ev, po;
  const float *a_wh, *a_ws, *a_bs;

  if (in_sizes[0] == 2 * N_EDGES) {
    // Sorted-pytree flatten: edge_index, edge_s, edge_v, params{attn,ek,ev,k,out,q,v}
    // with per-GVP keys sorted {bs,bsv,wh,ws,wsv,wv}, then x_s, x_v.
    eidx = (const int*)d_in[0]; edge_s = F(1); edge_v = F(2);
    a_bs = F(3); a_wh = F(4); a_ws = F(5);
    auto G = [&](int b) {
      P6 p; p.bs = F(b); p.bsv = F(b + 1); p.wh = F(b + 2);
      p.ws = F(b + 3); p.wsv = F(b + 4); p.wv = F(b + 5); return p;
    };
    ek = G(6); ev = G(12); k = G(18); po = G(24); q = G(30); v = G(36);
    x_s = F(42); x_v = F(43);
  } else {
    // Dict-insertion order: x_s, x_v, edge_s, edge_v,
    // params{q,k,v,ek,ev}(wh,ws,bs,wv,wsv,bsv), attn(wh,ws,bs), out, edge_index.
    x_s = F(0); x_v = F(1); edge_s = F(2); edge_v = F(3);
    auto G = [&](int b) {
      P6 p; p.wh = F(b); p.ws = F(b + 1); p.bs = F(b + 2);
      p.wv = F(b + 3); p.wsv = F(b + 4); p.bsv = F(b + 5); return p;
    };
    q = G(4); k = G(10); v = G(16); ek = G(22); ev = G(28);
    a_wh = F(34); a_ws = F(35); a_bs = F(36);
    po = G(37);
    eidx = (const int*)d_in[43];
  }
  const int* src = eidx;
  const int* dst = eidx + N_EDGES;

  float* w    = (float*)d_ws;
  float* qs   = w + OFF_QS;   float* qvv  = w + OFF_QV;
  float* ks   = w + OFF_KS;   float* kv   = w + OFF_KV;
  float* vs   = w + OFF_VS;   float* vv   = w + OFF_VV;
  float* en   = w + OFF_EN;
  float* nmax = w + OFF_NMAX; float* nden = w + OFF_NDEN;
  float* aggs = w + OFF_AGGS; float* aggv = w + OFF_AGGV;

  float* out_s = (float*)d_out;
  float* out_v = out_s + (size_t)N_NODES * 128;

  const int nodeBlocks = (N_NODES + 15) / 16;   // 625
  const int edgeBlocks = N_EDGES / 64;          // 5000

  init_kernel<<<(N_NODES * 128 + 255) / 256, 256, 0, stream>>>(nmax, nden, aggs, aggv);

  node_gvp_kernel<<<nodeBlocks, 128, 0, stream>>>(x_s, x_v, q.wh, q.ws, q.bs, q.wv,
                                                  q.wsv, q.bsv, qs, qvv, N_NODES);
  node_gvp_kernel<<<nodeBlocks, 128, 0, stream>>>(x_s, x_v, k.wh, k.ws, k.bs, k.wv,
                                                  k.wsv, k.bsv, ks, kv, N_NODES);
  node_gvp_kernel<<<nodeBlocks, 128, 0, stream>>>(x_s, x_v, v.wh, v.ws, v.bs, v.wv,
                                                  v.wsv, v.bsv, vs, vv, N_NODES);

  edge_kernel<0><<<edgeBlocks, 128, 0, stream>>>(
      edge_s, edge_v, src, dst, ek.wh, ek.ws, ek.bs, ek.wv, ek.wsv, ek.bsv,
      ks, kv, qs, qvv, a_wh, a_ws, a_bs, en, nmax, nullptr, nullptr, nullptr);

  softmax_kernel<<<(N_EDGES * 4 + 255) / 256, 256, 0, stream>>>(dst, en, nmax, nden);

  edge_kernel<1><<<edgeBlocks, 128, 0, stream>>>(
      edge_s, edge_v, src, dst, ev.wh, ev.ws, ev.bs, ev.wv, ev.wsv, ev.bsv,
      vs, vv, nullptr, nullptr, nullptr, nullptr, nullptr, en, nullptr, nden,
      aggs, aggv);

  node_gvp_kernel<<<nodeBlocks, 128, 0, stream>>>(aggs, aggv, po.wh, po.ws, po.bs,
                                                  po.wv, po.wsv, po.bsv, out_s, out_v,
                                                  N_NODES);
}
